// CARAFE_Downsample_3_kernelexp_56719338111492
// MI455X (gfx1250) — matrix operand order repaired and checked by reading, not compiled, not run on hardware
//
#include <hip/hip_runtime.h>
#include <hip/hip_bf16.h>

typedef __attribute__((ext_vector_type(16))) _Float16 v16h;
typedef __attribute__((ext_vector_type(8)))  float    v8f;

union Frag16 {
    v16h v;
    _Float16 h[16];
    unsigned int u[8];
};

#define HW 16384   // 128*128
#define HOWO 4096  // 64*64

// -------- async VMEM -> LDS helpers (CDNA5, tracked on ASYNCcnt) -----------
__device__ __forceinline__ void async_b32(void* lds, const void* gaddr) {
    asm volatile("global_load_async_to_lds_b32 %0, %1, off"
                 :: "v"((unsigned)(unsigned long long)lds), "v"(gaddr)
                 : "memory");
}
__device__ __forceinline__ void async_b128(void* lds, const void* gaddr) {
    asm volatile("global_load_async_to_lds_b128 %0, %1, off"
                 :: "v"((unsigned)(unsigned long long)lds), "v"(gaddr)
                 : "memory");
}
__device__ __forceinline__ void wait_async0() {
    asm volatile("s_wait_asynccnt 0" ::: "memory");
}

// ---------------------------------------------------------------------------
// Kernel 0: weight prep.
//  Wc  : f16 copy of w_comp, [64][256]   (row = out channel, contiguous K)
//  Wm  : f16 packed [48][576], rows 0..24 = w_enc, 25..40 = w_kenc, rest 0.
//        column k = (kh*3+kw)*64 + cc  (matches im2col order of k_maskconv)
// ---------------------------------------------------------------------------
__global__ __launch_bounds__(256) void k_prep(const float* __restrict__ w_comp,
                                              const float* __restrict__ w_enc,
                                              const float* __restrict__ w_kenc,
                                              _Float16* __restrict__ Wc,
                                              _Float16* __restrict__ Wm) {
    int i = blockIdx.x * 256 + threadIdx.x;
    if (i < 64 * 256) {
        Wc[i] = (_Float16)w_comp[i];
    }
    if (i < 48 * 576) {
        int n = i / 576, k = i % 576;
        int loc = k >> 6, cc = k & 63;
        int kh = loc / 3, kw = loc % 3;
        float v = 0.f;
        if (n < 25)      v = w_enc[((n * 64 + cc) * 3 + kh) * 3 + kw];
        else if (n < 41) v = w_kenc[(((n - 25) * 64 + cc) * 3 + kh) * 3 + kw];
        Wm[i] = (_Float16)v;
    }
}

// ---------------------------------------------------------------------------
// Kernel 1: channel compressor 1x1 conv as WMMA GEMM.
//  D[16px x 16cc] += A[16px x 32c] * B[32c x 16cc]
//  A: x (NCHW fp32) converted to f16 on the fly; lane m = pixel.
//  B: W^T from LDS (async-staged; [cc] rows, 129-dword pitch -> conflict-free)
//  Output: cx_t[p][64] f16 (pixel-major, 64 channels contiguous).
//  Wave = 16 pixels x 64 out-channels; block = 8 waves = 128 pixels.
// ---------------------------------------------------------------------------
__global__ __launch_bounds__(256) void k_compress(const float* __restrict__ x,
                                                  const unsigned int* __restrict__ Wc_u32,
                                                  const float* __restrict__ bias,
                                                  _Float16* __restrict__ cxt) {
    __shared__ unsigned int lw[64 * 129];
    for (int q = threadIdx.x; q < 64 * 128; q += 256) {
        int row = q >> 7, col = q & 127;
        async_b32(&lw[row * 129 + col], Wc_u32 + q);
    }
    wait_async0();
    __syncthreads();

    int wave = threadIdx.x >> 5;
    int lane = threadIdx.x & 31;
    int nl   = lane & 15;     // N column / M row selector
    int kh   = lane >> 4;     // half-lane K/M split

    int pbase = blockIdx.x * 128 + wave * 16;
    int p  = pbase + nl;      // this lane's A-row pixel
    int b  = p >> 14;
    int hw = p & 16383;
    const float* xb = x + (size_t)b * 256 * HW + hw;

    float bias4[4];
#pragma unroll
    for (int nt = 0; nt < 4; ++nt) bias4[nt] = bias[nt * 16 + nl];

    v8f acc[4] = {};

#pragma unroll
    for (int kt = 0; kt < 8; ++kt) {
        Frag16 a;
#pragma unroll
        for (int j = 0; j < 8; ++j) {
            int klocal = (j < 4) ? (2 * j + 8 * kh) : (16 + 2 * (j - 4) + 8 * kh);
            int c = kt * 32 + klocal;
            float f0 = xb[(size_t)c * HW];
            float f1 = xb[(size_t)(c + 1) * HW];
            a.h[2 * j]     = (_Float16)f0;
            a.h[2 * j + 1] = (_Float16)f1;
        }
#pragma unroll
        for (int nt = 0; nt < 4; ++nt) {
            Frag16 bf;
            int n = nt * 16 + nl;
#pragma unroll
            for (int j = 0; j < 8; ++j)
                bf.u[j] = lw[n * 129 + kt * 16 + ((j < 4) ? (j + 4 * kh) : (8 + (j - 4) + 4 * kh))];
            acc[nt] = __builtin_amdgcn_wmma_f32_16x16x32_f16(
                false, a.v, false, bf.v, (short)0, acc[nt], false, false);
        }
    }

    // D layout: VGPR r -> row m = r + 8*kh ; lane nl -> column n
#pragma unroll
    for (int nt = 0; nt < 4; ++nt) {
        int n = nt * 16 + nl;
#pragma unroll
        for (int r = 0; r < 8; ++r) {
            int mrow = r + 8 * kh;
            cxt[(size_t)(pbase + mrow) * 64 + n] = (_Float16)(acc[nt][r] + bias4[nt]);
        }
    }
}

// ---------------------------------------------------------------------------
// Kernel 2: fused mask(25) + ke(16) 3x3 stride-2 conv as WMMA im2col GEMM.
//  M = output pixel (16 consecutive wo per wave), K = 9*64 = 576, N = 48.
//  18 K-steps of 32 (2 per spatial tap), zero-fill halo via predicated loads.
//  Wm async-staged to LDS once per block (292-dword pitch: 16B aligned for
//  b128, 36*m mod 64 gives 16 distinct banks). Writes fp32 logits[p][48].
// ---------------------------------------------------------------------------
__global__ __launch_bounds__(256) void k_maskconv(const unsigned int* __restrict__ cxt_u32,
                                                  const unsigned int* __restrict__ Wm_u32,
                                                  const float* __restrict__ b_enc,
                                                  const float* __restrict__ b_kenc,
                                                  float* __restrict__ logits) {
    __shared__ unsigned int lwm[48 * 292];
    for (int q = threadIdx.x; q < 48 * 72; q += 256) {  // 16B chunks
        int row = q / 72, c4 = (q % 72) * 4;
        async_b128(&lwm[row * 292 + c4], Wm_u32 + (row * 288 + c4));
    }
    wait_async0();
    __syncthreads();

    int wave = threadIdx.x >> 5;
    int lane = threadIdx.x & 31;
    int nl   = lane & 15;
    int kh   = lane >> 4;

    int pbase = blockIdx.x * 128 + wave * 16;
    int p  = pbase + nl;
    int b  = p >> 12;
    int ho = (p >> 6) & 63;
    int wo = p & 63;

    v8f acc[3] = {};

#pragma unroll
    for (int t = 0; t < 18; ++t) {
        int loc = t >> 1;
        int dh = loc / 3, dw = loc % 3;
        int ccb = (t & 1) * 32;
        int ih = 2 * ho + dh - 1;
        int iw = 2 * wo + dw - 1;
        bool ok = ((unsigned)ih < 128u) && ((unsigned)iw < 128u);
        const unsigned int* src =
            cxt_u32 + (size_t)((b * 128 + ih) * 128 + iw) * 32;  // 32 dwords = 64 f16

        Frag16 a;
#pragma unroll
        for (int j = 0; j < 8; ++j) {
            int klocal = (j < 4) ? (2 * j + 8 * kh) : (16 + 2 * (j - 4) + 8 * kh);
            a.u[j] = ok ? src[(ccb + klocal) >> 1] : 0u;
        }
#pragma unroll
        for (int nt = 0; nt < 3; ++nt) {
            Frag16 bf;
            int n = nt * 16 + nl;
#pragma unroll
            for (int j = 0; j < 8; ++j)
                bf.u[j] = lwm[n * 292 + t * 16 + ((j < 4) ? (j + 4 * kh) : (8 + (j - 4) + 4 * kh))];
            acc[nt] = __builtin_amdgcn_wmma_f32_16x16x32_f16(
                false, a.v, false, bf.v, (short)0, acc[nt], false, false);
        }
    }

#pragma unroll
    for (int nt = 0; nt < 3; ++nt) {
        int n = nt * 16 + nl;
        float bb = 0.f;
        if (n < 25)      bb = b_enc[n];
        else if (n < 41) bb = b_kenc[n - 25];
#pragma unroll
        for (int r = 0; r < 8; ++r) {
            int mrow = r + 8 * kh;
            logits[(size_t)(pbase + mrow) * 48 + n] = acc[nt][r] + bb;
        }
    }
}

// ---------------------------------------------------------------------------
// Kernel 3: per-pixel temperature + softmax. mask written NCHW (4,25,64,64).
// ---------------------------------------------------------------------------
__global__ __launch_bounds__(256) void k_finalize(const float* __restrict__ logits,
                                                  float* __restrict__ maskf) {
    int p = blockIdx.x * 256 + threadIdx.x;  // 16384 pixels
    const float* L = logits + (size_t)p * 48;
    float prod = 1.f;
#pragma unroll
    for (int k = 0; k < 16; ++k) prod *= L[25 + k];
    float temp = fminf(10.f, fmaxf(-10.f, prod));

    float v[25];
    float mx = -3.4e38f;
#pragma unroll
    for (int k = 0; k < 25; ++k) {
        v[k] = L[k] * temp;
        mx = fmaxf(mx, v[k]);
    }
    float s = 0.f;
#pragma unroll
    for (int k = 0; k < 25; ++k) {
        v[k] = __expf(v[k] - mx);
        s += v[k];
    }
    float inv = 1.f / s;

    int b = p >> 12, ho = (p >> 6) & 63, wo = p & 63;
#pragma unroll
    for (int k = 0; k < 25; ++k)
        maskf[(((size_t)b * 25 + k) * 64 + ho) * 64 + wo] = v[k] * inv;
}

// ---------------------------------------------------------------------------
// Kernel 4: feature reassembly. Block = (b, 4-channel group, ho row).
//  Mask row (25x64) async-staged; 5 zero-padded x rows staged with predicated
//  sync path (needs zero fill). Thread = (channel-in-group, wo); 25 FMAs.
//  x is L2-resident (64MB < 192MB L2) so the 5x overlap re-reads hit L2.
// ---------------------------------------------------------------------------
__global__ __launch_bounds__(256) void k_reassemble(const float* __restrict__ x,
                                                    const float* __restrict__ maskf,
                                                    float* __restrict__ out) {
    __shared__ float xl[4][5][132];
    __shared__ float ml[25][64];

    int blk = blockIdx.x;           // ((b*64 + cg)*64 + ho)
    int ho = blk & 63;
    int cg = (blk >> 6) & 63;
    int b  = blk >> 12;

    for (int q = threadIdx.x; q < 25 * 64; q += 256) {
        int k = q >> 6, wo = q & 63;
        async_b32(&ml[k][wo], maskf + ((((size_t)b * 25 + k) * 64 + ho) * 64 + wo));
    }
    for (int q = threadIdx.x; q < 4 * 5 * 132; q += 256) {
        int cl = q / 660;
        int rem = q % 660;
        int r = rem / 132;
        int col = rem % 132;
        int ih = 2 * ho - 2 + r;
        int iw = col - 2;
        float v = 0.f;
        if ((unsigned)ih < 128u && (unsigned)iw < 128u) {
            int c = cg * 4 + cl;
            v = x[(((size_t)b * 256 + c) * 128 + ih) * 128 + iw];
        }
        xl[cl][r][col] = v;
    }
    wait_async0();
    __syncthreads();

    int wo = threadIdx.x & 63;
    int cl = threadIdx.x >> 6;
    float acc = 0.f;
#pragma unroll
    for (int k = 0; k < 25; ++k) {
        int khh = k / 5, kw = k % 5;
        acc += xl[cl][khh][2 * wo + kw] * ml[k][wo];
    }
    int c = cg * 4 + cl;
    out[(((size_t)b * 256 + c) * 64 + ho) * 64 + wo] = acc;
}

// ---------------------------------------------------------------------------
// Host-side launcher. Workspace layout (bytes, 256-aligned):
//   [0        , 8388608)  cx_t   f16 [65536][64]
//   [8388608  , 8421376)  Wc     f16 [64][256]
//   [8421376  , 8476672)  Wm     f16 [48][576]
//   [8476672  , 11622400) logits f32 [16384][48]
//   [11622400 , 13260800) maskf  f32 [4][25][64][64]
// ---------------------------------------------------------------------------
extern "C" void kernel_launch(void* const* d_in, const int* in_sizes, int n_in,
                              void* d_out, int out_size, void* d_ws, size_t ws_size,
                              hipStream_t stream) {
    const float* x      = (const float*)d_in[0];
    const float* w_comp = (const float*)d_in[1];
    const float* b_comp = (const float*)d_in[2];
    const float* w_enc  = (const float*)d_in[3];
    const float* b_enc  = (const float*)d_in[4];
    const float* w_kenc = (const float*)d_in[5];
    const float* b_kenc = (const float*)d_in[6];
    float* out = (float*)d_out;

    char* ws = (char*)d_ws;
    _Float16*     cxt    = (_Float16*)(ws + 0);
    _Float16*     Wc     = (_Float16*)(ws + 8388608);
    _Float16*     Wm     = (_Float16*)(ws + 8421376);
    float*        logits = (float*)(ws + 8476672);
    float*        maskf  = (float*)(ws + 11622400);

    k_prep<<<108, 256, 0, stream>>>(w_comp, w_enc, w_kenc, Wc, Wm);
    k_compress<<<512, 256, 0, stream>>>(x, (const unsigned int*)Wc, b_comp, cxt);
    k_maskconv<<<128, 256, 0, stream>>>((const unsigned int*)cxt,
                                        (const unsigned int*)Wm, b_enc, b_kenc, logits);
    k_finalize<<<64, 256, 0, stream>>>(logits, maskf);
    k_reassemble<<<16384, 256, 0, stream>>>(x, maskf, out);
}
